// CausalSelfAttention_9242769622520
// MI455X (gfx1250) — matrix operand order repaired and checked by reading, not compiled, not run on hardware
//
#include <hip/hip_runtime.h>
#include <hip/hip_bf16.h>
#include <math.h>
#include <cstdint>

// ---------------------------------------------------------------------------
// CausalSelfAttention on MI455X (gfx1250), wave32 + v_wmma_f32_16x16x32_bf16,
// GLOBAL_LOAD_ASYNC_TO_LDS_B128 double-buffered weight staging.
// B=2, S=2048, H=16, D=64, HID=1024.  fp32 I/O, bf16 matmuls, fp32 accum.
// ---------------------------------------------------------------------------

constexpr int B_  = 2;
constexpr int S_  = 2048;
constexpr int H_  = 16;
constexpr int D_  = 64;
constexpr int HID_ = H_ * D_;     // 1024
constexpr int NTOK_ = B_ * S_;    // 4096

typedef __attribute__((ext_vector_type(16))) __bf16 v16bf;
typedef __attribute__((ext_vector_type(8)))  __bf16 v8bf;
typedef __attribute__((ext_vector_type(8)))  float  v8f;

__device__ __forceinline__ v8f wmma_bf16(v16bf a, v16bf b, v8f c) {
  // D = A(16x32 bf16) * B(32x16 bf16) + C(16x16 f32)
  return __builtin_amdgcn_wmma_f32_16x16x32_bf16(
      /*neg_a=*/false, a, /*neg_b=*/false, b,
      /*c_mod=*/(short)0, c, /*reuse_a=*/false, /*reuse_b=*/false);
}

// A-fragment (16x32, bf16): lane holds row (lane&15); with half = lane>>4 the
// 16 elements are K = half*16 + [0..15], contiguous.
// Caller passes p = rowPtr + k0 + half*16.
__device__ __forceinline__ v16bf load_frag_a(const __bf16* p) {
  v16bf a;
#pragma unroll
  for (int i = 0; i < 16; ++i) a[i] = p[i];
  return a;
}

// B-fragment (32x16, bf16): lane holds column (lane&15); elements are
// K = half*16 + [0..15], contiguous. Caller passes p = colPtr + k0 + half*16.
__device__ __forceinline__ v16bf load_frag_b(const __bf16* p) {
  v16bf b;
#pragma unroll
  for (int i = 0; i < 16; ++i) b[i] = p[i];
  return b;
}

// ---------------------------------------------------------------------------
// fp32 -> bf16, 8 elements per thread (b128 in/out)
// ---------------------------------------------------------------------------
__global__ void f32_to_bf16_kernel(const float* __restrict__ in,
                                   __bf16* __restrict__ out, int n8) {
  int i = blockIdx.x * blockDim.x + threadIdx.x;
  if (i < n8) {
    const float4* p = (const float4*)in + (size_t)i * 2;
    float4 a = p[0], b = p[1];
    v8bf o;
    o[0] = (__bf16)a.x; o[1] = (__bf16)a.y; o[2] = (__bf16)a.z; o[3] = (__bf16)a.w;
    o[4] = (__bf16)b.x; o[5] = (__bf16)b.y; o[6] = (__bf16)b.z; o[7] = (__bf16)b.w;
    ((v8bf*)out)[i] = o;
  }
}

// ---------------------------------------------------------------------------
// GEMM: out = A[M][K] * W[N][K]^T + bias[N]
// Block = 4 waves = 64(M) x 64(N); wave owns 16(M) x 64(N).
// W tile (64x32 bf16, 4KB) staged in LDS via async copy, double buffered,
// shared by all 4 waves. A fragments loaded directly from global.
// mode 0: fp32 out[M][N]
// mode 1: bf16 out[b][h][s][d]   (Q, K)
// mode 2: bf16 out[b][h][d][s]   (V transposed for PV WMMA B-fragments)
// ---------------------------------------------------------------------------
__global__ __launch_bounds__(128) void gemm_bf16_wmma(
    const __bf16* __restrict__ A, const __bf16* __restrict__ W,
    const float* __restrict__ bias, void* __restrict__ out,
    int M, int N, int K, int mode) {
  __shared__ __bf16 sB[2][64][36];   // padded rows (72B) -> conflict-free frags

  const int t    = threadIdx.x;
  const int wid  = t >> 5;
  const int half = (t >> 4) & 1;
  const int lm   = t & 15;
  const int m0 = (blockIdx.y * 4 + wid) * 16;
  const int n0 = blockIdx.x * 64;

  // Async-copy mapping: thread t moves 32 bytes of the 64x32 W tile:
  // relative row rn = t>>1, k offset koff = (t&1)*16.
  const int rn   = t >> 1;
  const int koff = (t & 1) * 16;
  const __bf16* gsrc = W + (size_t)(n0 + rn) * K + koff;
  const uint32_t lds0 = (uint32_t)(uintptr_t)&sB[0][rn][koff];
  const uint32_t ldsStride =
      (uint32_t)((uintptr_t)&sB[1][0][0] - (uintptr_t)&sB[0][0][0]);

  auto issue_stage = [&](int st, int k0) {
    uint64_t ga = (uint64_t)(uintptr_t)(gsrc + k0);
    uint32_t la = lds0 + (uint32_t)st * ldsStride;
    asm volatile(
        "global_load_async_to_lds_b128 %0, %1, off\n\t"
        "global_load_async_to_lds_b128 %0, %1, off offset:16"
        :: "v"(la), "v"(ga) : "memory");
  };

  v8f acc[4] = {};
  issue_stage(0, 0);

  const int KT = K >> 5;
  const __bf16* aRow = A + (size_t)(m0 + lm) * K + half * 16;
  for (int kt = 0; kt < KT; ++kt) {
    const int st = kt & 1;
    asm volatile("s_wait_asynccnt 0" ::: "memory");
    __syncthreads();                       // stage st fully in LDS

    v16bf a = load_frag_a(aRow + (kt << 5));
    v16bf bfr[4];
#pragma unroll
    for (int c = 0; c < 4; ++c)
      bfr[c] = load_frag_b(&sB[st][c * 16 + lm][half * 16]);

    asm volatile("s_wait_dscnt 0" ::: "memory");
    __syncthreads();                       // all waves done reading stage st
    if (kt + 1 < KT) issue_stage(st ^ 1, (kt + 1) << 5);  // DMA overlaps WMMAs

#pragma unroll
    for (int c = 0; c < 4; ++c) acc[c] = wmma_bf16(a, bfr[c], acc[c]);
  }

#pragma unroll
  for (int c = 0; c < 4; ++c) {
    const int col = n0 + c * 16 + lm;
    const float bv = bias[col];
#pragma unroll
    for (int r = 0; r < 8; ++r) {
      const int row = m0 + r + 8 * half;
      const float v = acc[c][r] + bv;
      if (mode == 0) {
        ((float*)out)[(size_t)row * N + col] = v;
      } else {
        const int bb = row / S_, ss = row % S_;
        const int hh = col / D_, dd = col % D_;
        size_t idx;
        if (mode == 1)
          idx = (((size_t)bb * H_ + hh) * S_ + ss) * D_ + dd;
        else
          idx = (((size_t)bb * H_ + hh) * D_ + dd) * S_ + ss;
        ((__bf16*)out)[idx] = (__bf16)v;
      }
    }
  }
}

// ---------------------------------------------------------------------------
// Flash attention forward. One wave handles one (b,h, 16-query tile).
// Q,K: bf16 [b,h,s,d].  Vt: bf16 [b,h,d,s].  O: bf16 [b,s,hid].
// 32 keys per iteration: scores via 4 WMMAs (K-dim = D = 64, two steps of 32),
// softmax bookkeeping through LDS, P*V via 4 WMMAs (K-dim = 32 keys).
// V fragments are loaded early so global latency hides under the softmax.
// ---------------------------------------------------------------------------
#define AW 4  // waves per block
#define NEG_INF (-3.0e38f)

__global__ __launch_bounds__(128) void attn_fwd_wmma(
    const __bf16* __restrict__ Q, const __bf16* __restrict__ Kmat,
    const __bf16* __restrict__ Vt, __bf16* __restrict__ O) {
  __shared__ float  sS[AW][16][32];   // raw scores
  __shared__ __bf16 sP[AW][16][32];   // exp(scores) in bf16
  __shared__ float  sM[AW][16];       // running max
  __shared__ float  sL[AW][16];       // running denom
  __shared__ float  sF[AW][16];       // rescale factor this block

  const int lane = threadIdx.x & 31;
  const int w    = threadIdx.x >> 5;
  const int half = lane >> 4;
  const int lm   = lane & 15;

  const int tile = blockIdx.x * AW + w;     // 0 .. B*H*(S/16)-1
  const int qt = tile & (S_ / 16 - 1);      // S/16 = 128 tiles per head
  const int bh = tile >> 7;
  const int q0 = qt * 16;

  const __bf16* Qp = Q    + (size_t)bh * S_ * D_;
  const __bf16* Kp = Kmat + (size_t)bh * S_ * D_;
  const __bf16* Vp = Vt   + (size_t)bh * D_ * S_;

  // Q fragments for the whole tile (K-dim = D = 64 -> two 16x32 fragments)
  const v16bf qa0 = load_frag_a(Qp + (size_t)(q0 + lm) * D_ + 0  + half * 16);
  const v16bf qa1 = load_frag_a(Qp + (size_t)(q0 + lm) * D_ + 32 + half * 16);

  v8f o[4] = {};
  if (lane < 16) { sM[w][lane] = NEG_INF; sL[w][lane] = 0.f; }
  asm volatile("s_wait_dscnt 0" ::: "memory");

  const int kend = q0 + 16;  // need keys <= q0+15 (causal)
  for (int kb0 = 0; kb0 < kend; kb0 += 32) {
    // ---- scores S = (Q K^T) * 1/sqrt(D), two 16x16 column tiles ----
    v8f s0 = {}, s1 = {};
    {
      v16bf b00 = load_frag_b(Kp + (size_t)(kb0 + lm) * D_ + 0  + half * 16);
      v16bf b01 = load_frag_b(Kp + (size_t)(kb0 + lm) * D_ + 32 + half * 16);
      s0 = wmma_bf16(qa0, b00, s0);
      s0 = wmma_bf16(qa1, b01, s0);
      v16bf b10 = load_frag_b(Kp + (size_t)(kb0 + 16 + lm) * D_ + 0  + half * 16);
      v16bf b11 = load_frag_b(Kp + (size_t)(kb0 + 16 + lm) * D_ + 32 + half * 16);
      s1 = wmma_bf16(qa0, b10, s1);
      s1 = wmma_bf16(qa1, b11, s1);
    }

    // ---- start V fragment loads now; consumed after softmax ----
    v16bf vb[4];
#pragma unroll
    for (int c = 0; c < 4; ++c)
      vb[c] = load_frag_b(Vp + (size_t)(c * 16 + lm) * S_ + kb0 + half * 16);

#pragma unroll
    for (int r = 0; r < 8; ++r) {
      sS[w][r + 8 * half][lm]      = s0[r] * 0.125f;   // 1/sqrt(64)
      sS[w][r + 8 * half][lm + 16] = s1[r] * 0.125f;
    }
    asm volatile("s_wait_dscnt 0" ::: "memory");

    // ---- per-row masked online softmax (lanes 0..15, one row each) ----
    if (lane < 16) {
      const int row = lane;
      const int qg  = q0 + row;
      const float mold = sM[w][row];
      float mmax = mold;
      float sc[32];
#pragma unroll
      for (int j = 0; j < 32; ++j) {
        float x = sS[w][row][j];
        if (kb0 + j > qg) x = NEG_INF;        // causal mask
        sc[j] = x;
        mmax = fmaxf(mmax, x);
      }
      const float fac = __expf(mold - mmax);  // 0 on first block
      float lsum = 0.f;
#pragma unroll
      for (int j = 0; j < 32; ++j) {
        const float p = __expf(sc[j] - mmax);
        sP[w][row][j] = (__bf16)p;
        lsum += p;
      }
      sM[w][row] = mmax;
      sL[w][row] = sL[w][row] * fac + lsum;
      sF[w][row] = fac;
    }
    asm volatile("s_wait_dscnt 0" ::: "memory");

    // ---- rescale accumulators ----
#pragma unroll
    for (int r = 0; r < 8; ++r) {
      const float f = sF[w][r + 8 * half];
#pragma unroll
      for (int c = 0; c < 4; ++c) o[c][r] *= f;
    }

    // ---- O += P * V  (A = P from LDS, B = V^T columns = contiguous keys) ----
    v16bf pa = load_frag_a(&sP[w][lm][half * 16]);
#pragma unroll
    for (int c = 0; c < 4; ++c) o[c] = wmma_bf16(pa, vb[c], o[c]);
  }

  // ---- normalize and write O as bf16 [b][s][h*64+d] for the proj GEMM ----
  const int bb = bh / H_, hh = bh % H_;
#pragma unroll
  for (int r = 0; r < 8; ++r) {
    const int qrow = q0 + r + 8 * half;
    const float inv = 1.f / sL[w][r + 8 * half];
    const size_t base = ((size_t)bb * S_ + qrow) * HID_ + hh * D_;
#pragma unroll
    for (int c = 0; c < 4; ++c)
      O[base + c * 16 + lm] = (__bf16)(o[c][r] * inv);
  }
}

// ---------------------------------------------------------------------------
// Host-side launch
// ---------------------------------------------------------------------------
extern "C" void kernel_launch(void* const* d_in, const int* in_sizes, int n_in,
                              void* d_out, int out_size, void* d_ws, size_t ws_size,
                              hipStream_t stream) {
  const float* x  = (const float*)d_in[0];
  const float* Wq = (const float*)d_in[1];
  const float* bq = (const float*)d_in[2];
  const float* Wk = (const float*)d_in[3];
  const float* bk = (const float*)d_in[4];
  const float* Wv = (const float*)d_in[5];
  const float* bv = (const float*)d_in[6];
  const float* Wp = (const float*)d_in[7];
  const float* bp = (const float*)d_in[8];

  // Workspace carve-up (48 MiB total)
  char* ws = (char*)d_ws;
  const size_t nX = (size_t)NTOK_ * HID_;   // 4 Mi elems
  const size_t nW = (size_t)HID_ * HID_;    // 1 Mi elems
  __bf16* xbf = (__bf16*)ws;  ws += nX * sizeof(__bf16);
  __bf16* wqb = (__bf16*)ws;  ws += nW * sizeof(__bf16);
  __bf16* wkb = (__bf16*)ws;  ws += nW * sizeof(__bf16);
  __bf16* wvb = (__bf16*)ws;  ws += nW * sizeof(__bf16);
  __bf16* wpb = (__bf16*)ws;  ws += nW * sizeof(__bf16);
  __bf16* qb  = (__bf16*)ws;  ws += nX * sizeof(__bf16);
  __bf16* kb  = (__bf16*)ws;  ws += nX * sizeof(__bf16);
  __bf16* vtb = (__bf16*)ws;  ws += nX * sizeof(__bf16);
  __bf16* ob  = (__bf16*)ws;  ws += nX * sizeof(__bf16);

  // 1) bf16 conversions (8 elems / thread)
  f32_to_bf16_kernel<<<(int)((nX / 8 + 255) / 256), 256, 0, stream>>>(x,  xbf, (int)(nX / 8));
  f32_to_bf16_kernel<<<(int)((nW / 8 + 255) / 256), 256, 0, stream>>>(Wq, wqb, (int)(nW / 8));
  f32_to_bf16_kernel<<<(int)((nW / 8 + 255) / 256), 256, 0, stream>>>(Wk, wkb, (int)(nW / 8));
  f32_to_bf16_kernel<<<(int)((nW / 8 + 255) / 256), 256, 0, stream>>>(Wv, wvb, (int)(nW / 8));
  f32_to_bf16_kernel<<<(int)((nW / 8 + 255) / 256), 256, 0, stream>>>(Wp, wpb, (int)(nW / 8));

  // 2) Q/K/V projections (Q,K -> [b,h,s,d] bf16; V -> [b,h,d,s] bf16)
  dim3 gg(HID_ / 64, NTOK_ / 64);  // (16, 64)
  gemm_bf16_wmma<<<gg, 128, 0, stream>>>(xbf, wqb, bq, qb,  NTOK_, HID_, HID_, 1);
  gemm_bf16_wmma<<<gg, 128, 0, stream>>>(xbf, wkb, bk, kb,  NTOK_, HID_, HID_, 1);
  gemm_bf16_wmma<<<gg, 128, 0, stream>>>(xbf, wvb, bv, vtb, NTOK_, HID_, HID_, 2);

  // 3) flash attention: B*H*(S/16) = 4096 waves, 4 waves/block
  const int ntiles = B_ * H_ * (S_ / 16);
  attn_fwd_wmma<<<ntiles / AW, 32 * AW, 0, stream>>>(qb, kb, vtb, ob);

  // 4) output projection -> fp32 d_out
  gemm_bf16_wmma<<<gg, 128, 0, stream>>>(ob, wpb, bp, d_out, NTOK_, HID_, HID_, 0);
}
